// TABlock_326417515017
// MI455X (gfx1250) — matrix operand order repaired and checked by reading, not compiled, not run on hardware
//
#include <hip/hip_runtime.h>
#include <hip/hip_bf16.h>

typedef __bf16 bf16;
typedef __bf16 bf16x16 __attribute__((ext_vector_type(16)));
typedef float  v8f     __attribute__((ext_vector_type(8)));
typedef int    v4i     __attribute__((ext_vector_type(4)));

#ifndef __has_builtin
#define __has_builtin(x) 0
#endif
#if __has_builtin(__builtin_amdgcn_global_load_async_to_lds_b128) && \
    __has_builtin(__builtin_amdgcn_s_wait_asynccnt)
#define USE_ASYNC_LDS 1
#else
#define USE_ASYNC_LDS 0
#endif

// Address-space helpers: generic->AS1 (global, 64-bit) and generic->AS3 (LDS
// offset = low 32 bits of the generic address per the CDNA5 aperture rules).
// The async builtin takes int4 addrspace(1)* / int4 addrspace(3)* operands.
typedef __attribute__((address_space(1))) v4i gv4i;
typedef __attribute__((address_space(3))) v4i lv4i;
#define TO_GLOBAL(p) ((gv4i*)(unsigned long long)(const void*)(p))
#define TO_LDS(p)    ((lv4i*)(unsigned int)(unsigned long long)(const void*)(p))

union FragU { bf16x16 v; uint4 q[2]; };

// A fragment: 16x32 (MxK) bf16, source row-major [m][k], stride ld elements.
// Lane L: row m = L%16; k values {k0..k0+7} U {k0+16..k0+23}, k0 = (L<16?0:8).
__device__ __forceinline__ bf16x16 load_frag_a(const bf16* base, int ld, int lane) {
    int m  = lane & 15;
    int k0 = (lane < 16) ? 0 : 8;
    const bf16* p = base + (size_t)m * ld + k0;
    FragU f;
    f.q[0] = *(const uint4*)p;
    f.q[1] = *(const uint4*)(p + 16);
    return f.v;
}

// B fragment: 32x16 (KxN) bf16, source stored N-major: [n][k], stride ld elements.
// Lane L: col n = L%16; k values {k0..k0+7} U {k0+16..k0+23}, k0 = (L<16?0:8).
__device__ __forceinline__ bf16x16 load_frag_b(const bf16* base, int ld, int lane) {
    int n  = lane & 15;
    int k0 = (lane < 16) ? 0 : 8;
    const bf16* p = base + (size_t)n * ld + k0;
    FragU f;
    f.q[0] = *(const uint4*)p;
    f.q[1] = *(const uint4*)(p + 16);
    return f.v;
}

__device__ __forceinline__ v8f wmma_bf(bf16x16 a, bf16x16 b, v8f c) {
    return __builtin_amdgcn_wmma_f32_16x16x32_bf16(false, a, false, b, (short)0, c, false, false);
}

__device__ __forceinline__ unsigned short bf16_bits(float x) {
    union { __bf16 h; unsigned short u; } cv; cv.h = (__bf16)x; return cv.u;
}
__device__ __forceinline__ float bits_bf16(unsigned short u) {
    union { __bf16 h; unsigned short u; } cv; cv.u = u; return (float)cv.h;
}

// ---------------------------------------------------------------------------
// Small helper kernels
// ---------------------------------------------------------------------------
__global__ void cast_kernel(const float* __restrict__ in, bf16* __restrict__ out, long n) {
    long i = (long)blockIdx.x * 256 + threadIdx.x;
    if (i < n) out[i] = (bf16)in[i];
}

// w: [Co][Ci][taps] fp32  ->  out: [tap][Co][Ci] bf16
__global__ void pack_w_kernel(const float* __restrict__ w, bf16* __restrict__ out,
                              int Co, int Ci, int taps, long n) {
    long i = (long)blockIdx.x * 256 + threadIdx.x;
    if (i >= n) return;
    int ci = (int)(i % Ci);
    long t1 = i / Ci;
    int co = (int)(t1 % Co);
    int tap = (int)(t1 / Co);
    out[i] = (bf16)w[((size_t)co * Ci + ci) * taps + tap];
}

// bn: [4][C] (gamma,beta,mean,var) -> scale/shift
__global__ void bn_prep_kernel(const float* __restrict__ bn,
                               float* __restrict__ scale, float* __restrict__ shift, int C) {
    int c = blockIdx.x * 256 + threadIdx.x;
    if (c >= C) return;
    float g = bn[c], b = bn[C + c], m = bn[2 * C + c], v = bn[3 * C + c];
    float s = g * rsqrtf(v + 1e-5f);
    scale[c] = s;
    shift[c] = b - m * s;
}

// [b][C][P] -> [b][P][C]   (bf16, C,P multiples of 32)
__global__ void transpose_kernel(const bf16* __restrict__ in, bf16* __restrict__ out,
                                 int C, int P) {
    __shared__ bf16 t[32][33];
    int b  = blockIdx.z;
    int p0 = blockIdx.x * 32, c0 = blockIdx.y * 32;
    int tx = threadIdx.x & 31, ty = threadIdx.x >> 5;
    const bf16* src = in  + (size_t)b * C * P;
    bf16*       dst = out + (size_t)b * C * P;
    #pragma unroll
    for (int i = 0; i < 32; i += 8) t[ty + i][tx] = src[(size_t)(c0 + ty + i) * P + p0 + tx];
    __syncthreads();
    #pragma unroll
    for (int i = 0; i < 32; i += 8) dst[(size_t)(p0 + ty + i) * C + c0 + tx] = t[tx][ty + i];
}

__global__ void add3_kernel(const bf16* __restrict__ a, const bf16* __restrict__ b,
                            const bf16* __restrict__ c, bf16* __restrict__ out, long n) {
    long i = (long)blockIdx.x * 256 + threadIdx.x;
    if (i < n) out[i] = (bf16)((float)a[i] + (float)b[i] + (float)c[i]);
}

// ---------------------------------------------------------------------------
// Generic implicit-GEMM conv kernel (WMMA bf16).
//   out[b][co][y*64+x] = epilogue( sum_{tap,ci} A[tap][co][ci] * X[b][ci][(y+ky)*64+(x+kx)] )
// TAPS = 9 (3x3, zero pad 1) or 1 (1x1 GEMM).
// Grid: (B*64 rows, ceil(Co/128)); block 256 (8 waves, 4co x 2pix of 32x32).
// ---------------------------------------------------------------------------
constexpr int EP_BN_RELU   = 0;
constexpr int EP_BIAS      = 1;
constexpr int EP_BIAS_LFM  = 2;
constexpr int EP_RES_ALPHA = 3;
constexpr int EP_BIAS_F32  = 4;

template <int TAPS, int EPI>
__global__ __launch_bounds__(256)
void conv_gemm_kernel(const bf16* __restrict__ A, size_t aStride,
                      const bf16* __restrict__ X,
                      bf16* __restrict__ Ybf, float* __restrict__ Yf,
                      const float* __restrict__ scale, const float* __restrict__ shift,
                      const float* __restrict__ bias,  const float* __restrict__ lfm,
                      const bf16* __restrict__ residual, const float* __restrict__ alphap,
                      int Co, int Ci) {
    __shared__ bf16 ldsA[128 * 32];
    __shared__ bf16 ldsB[64 * 32];
    const int HW = 4096;
    int tid = threadIdx.x, wave = tid >> 5, lane = tid & 31;
    int b = blockIdx.x >> 6, y = blockIdx.x & 63;
    int coBlk = blockIdx.y * 128;
    int wco = (wave >> 1) * 32, wpx = (wave & 1) * 32;
    v8f z = {0.f, 0.f, 0.f, 0.f, 0.f, 0.f, 0.f, 0.f};
    v8f acc[2][2];
    acc[0][0] = z; acc[0][1] = z; acc[1][0] = z; acc[1][1] = z;
    const bf16* Ab = A + (size_t)b * aStride;

    for (int tap = 0; tap < TAPS; ++tap) {
        int ky = (TAPS == 9) ? (tap / 3 - 1) : 0;
        int kx = (TAPS == 9) ? (tap % 3 - 1) : 0;
        int yy = y + ky;
        bool rowOk = (yy >= 0) && (yy < 64);
        const bf16* wTap = Ab + (size_t)tap * Co * Ci;
        for (int kc = 0; kc < Ci; kc += 32) {
            // --- A tile: 128 co x 32 k, row-major in LDS ---
            {
                int r = tid >> 1, h = (tid & 1) * 16;
                int co = coBlk + r;
                bf16* dstLds = &ldsA[r * 32 + h];
#if USE_ASYNC_LDS
                if (co < Co) {
                    const bf16* src = wTap + (size_t)co * Ci + kc + h;
                    __builtin_amdgcn_global_load_async_to_lds_b128(TO_GLOBAL(src),     TO_LDS(dstLds),     0, 0);
                    __builtin_amdgcn_global_load_async_to_lds_b128(TO_GLOBAL(src + 8), TO_LDS(dstLds + 8), 0, 0);
                    if (kc + 32 < Ci) __builtin_prefetch(wTap + (size_t)co * Ci + kc + 32 + h, 0, 1);
                } else {
                    uint4 zz = {0u, 0u, 0u, 0u};
                    ((uint4*)dstLds)[0] = zz;
                    ((uint4*)dstLds)[1] = zz;
                }
#else
                uint4 v0 = {0u, 0u, 0u, 0u}, v1 = v0;
                if (co < Co) {
                    const uint4* s = (const uint4*)(wTap + (size_t)co * Ci + kc + h);
                    v0 = s[0]; v1 = s[1];
                    if (kc + 32 < Ci) __builtin_prefetch(wTap + (size_t)co * Ci + kc + 32 + h, 0, 1);
                }
                uint4* d = (uint4*)dstLds;
                d[0] = v0; d[1] = v1;
#endif
            }
            // --- B tile: stored N-major [pix][k] (transposed on the fly) ---
            {
                int k = tid >> 3, seg = tid & 7;
                const bf16* s = X + ((size_t)b * Ci + kc + k) * HW + (size_t)yy * 64;
                if constexpr (TAPS == 1) {
                    // no halo: one b128 load covers this thread's 8 pixels
                    union { uint4 q; unsigned short us[8]; } vv;
                    vv.q = *(const uint4*)(s + seg * 8);
                    unsigned short* lb = (unsigned short*)ldsB;
                    #pragma unroll
                    for (int j = 0; j < 8; ++j) {
                        int px = seg * 8 + j;
                        lb[px * 32 + k] = vv.us[j];
                    }
                } else {
                    #pragma unroll
                    for (int j = 0; j < 8; ++j) {
                        int px = seg * 8 + j;
                        int xx = px + kx;
                        bf16 val = (bf16)0.f;
                        if (rowOk && xx >= 0 && xx < 64) val = s[xx];
                        ldsB[px * 32 + k] = val;
                    }
                }
            }
#if USE_ASYNC_LDS
            __builtin_amdgcn_s_wait_asynccnt(0);
#endif
            __syncthreads();
            bf16x16 a0 = load_frag_a(&ldsA[(wco + 0)  * 32], 32, lane);
            bf16x16 a1 = load_frag_a(&ldsA[(wco + 16) * 32], 32, lane);
            bf16x16 b0 = load_frag_b(&ldsB[(wpx + 0)  * 32], 32, lane);
            bf16x16 b1 = load_frag_b(&ldsB[(wpx + 16) * 32], 32, lane);
            acc[0][0] = wmma_bf(a0, b0, acc[0][0]);
            acc[0][1] = wmma_bf(a0, b1, acc[0][1]);
            acc[1][0] = wmma_bf(a1, b0, acc[1][0]);
            acc[1][1] = wmma_bf(a1, b1, acc[1][1]);
            __syncthreads();
        }
    }

    int nloc = lane & 15, moff = (lane < 16) ? 0 : 8;
    #pragma unroll
    for (int ti = 0; ti < 2; ++ti)
        #pragma unroll
        for (int tj = 0; tj < 2; ++tj) {
            int p = y * 64 + wpx + tj * 16 + nloc;
            #pragma unroll
            for (int r = 0; r < 8; ++r) {
                int c = coBlk + wco + ti * 16 + moff + r;
                if (c < Co) {
                    float v = acc[ti][tj][r];
                    size_t oidx = ((size_t)b * Co + c) * (size_t)HW + p;
                    if constexpr (EPI == EP_BN_RELU) {
                        v = fmaxf(v * scale[c] + shift[c], 0.f);
                        Ybf[oidx] = (bf16)v;
                    } else if constexpr (EPI == EP_BIAS) {
                        v += bias[c];
                        Ybf[oidx] = (bf16)v;
                    } else if constexpr (EPI == EP_BIAS_LFM) {
                        v = (v + bias[c]) * lfm[p];
                        Ybf[oidx] = (bf16)v;
                    } else if constexpr (EPI == EP_RES_ALPHA) {
                        v = alphap[0] * v + (float)residual[oidx];
                        Ybf[oidx] = (bf16)v;
                    } else {
                        v += bias[c];
                        Yf[oidx] = v;
                    }
                }
            }
        }
}

// ---------------------------------------------------------------------------
// PAM attention: one workgroup = 16 queries. Full 16x4096 score row block in
// dynamic LDS (bf16, 128KB). out = alpha * softmax(QK^T) V^T + xres.
//   qT,kT: [b][4096][64] bf16; vmat,xres,out: [b][512][4096] bf16.
// Grid: B*256, block 256, shmem 16*4096*2.
// ---------------------------------------------------------------------------
__global__ __launch_bounds__(256)
void pam_attention_kernel(const bf16* __restrict__ qT, const bf16* __restrict__ kT,
                          const bf16* __restrict__ vmat, const bf16* __restrict__ xres,
                          bf16* __restrict__ out, const float* __restrict__ alphap) {
    extern __shared__ char smem[];
    bf16* S = (bf16*)smem;                 // [16][4096]
    __shared__ bf16  qtile[16 * 64];
    __shared__ float wmaxs[8][16];
    __shared__ float rowmax[16];
    __shared__ float rowsum[16];

    int tid = threadIdx.x, wave = tid >> 5, lane = tid & 31;
    int bq = blockIdx.x;
    int b = bq >> 8, q0 = (bq & 255) * 16;
    int nloc = lane & 15, moff = (lane < 16) ? 0 : 8;

    // stage Q tile (16 queries x 64 channels)
    {
        int m = tid >> 4, cs = (tid & 15) * 4;
        *(uint2*)&qtile[m * 64 + cs] = *(const uint2*)&qT[((size_t)b * 4096 + q0 + m) * 64 + cs];
    }
    if (tid < 16) rowsum[tid] = 0.f;
    __syncthreads();

    bf16x16 aq0 = load_frag_a(qtile + 0,  64, lane);
    bf16x16 aq1 = load_frag_a(qtile + 32, 64, lane);

    // Phase 1: S = Q K^T (each wave covers key tiles wave, wave+8, ...)
    float lmax[8];
    #pragma unroll
    for (int r = 0; r < 8; ++r) lmax[r] = -3.0e38f;
    for (int i = 0; i < 32; ++i) {
        int kb = (wave + i * 8) * 16;
        v8f c = {0.f, 0.f, 0.f, 0.f, 0.f, 0.f, 0.f, 0.f};
        bf16x16 bb0 = load_frag_b(kT + ((size_t)b * 4096 + kb) * 64 + 0,  64, lane);
        bf16x16 bb1 = load_frag_b(kT + ((size_t)b * 4096 + kb) * 64 + 32, 64, lane);
        c = wmma_bf(aq0, bb0, c);
        c = wmma_bf(aq1, bb1, c);
        #pragma unroll
        for (int r = 0; r < 8; ++r) {
            float val = c[r];
            lmax[r] = fmaxf(lmax[r], val);
            S[(size_t)(moff + r) * 4096 + kb + nloc] = (bf16)val;
        }
    }
    // reduce max within each 16-lane half, then across waves via LDS
    #pragma unroll
    for (int m = 1; m < 16; m <<= 1) {
        #pragma unroll
        for (int r = 0; r < 8; ++r) lmax[r] = fmaxf(lmax[r], __shfl_xor(lmax[r], m, 32));
    }
    if (lane == 0) {
        #pragma unroll
        for (int r = 0; r < 8; ++r) wmaxs[wave][r] = lmax[r];
    }
    if (lane == 16) {
        #pragma unroll
        for (int r = 0; r < 8; ++r) wmaxs[wave][8 + r] = lmax[r];
    }
    __syncthreads();
    if (tid < 16) {
        float mv = wmaxs[0][tid];
        #pragma unroll
        for (int w = 1; w < 8; ++w) mv = fmaxf(mv, wmaxs[w][tid]);
        rowmax[tid] = mv;
    }
    __syncthreads();

    // Phase 2: S <- exp(S - rowmax), accumulate row sums
    {
        int row = tid >> 4, sub = tid & 15;
        float mv = rowmax[row], ps = 0.f;
        for (int j = sub; j < 4096; j += 16) {
            float e = __expf((float)S[(size_t)row * 4096 + j] - mv);
            S[(size_t)row * 4096 + j] = (bf16)e;
            ps += e;
        }
        atomicAdd(&rowsum[row], ps);
    }
    __syncthreads();

    // Phase 3: O = S V^T  (wave w covers output channels [64w, 64w+64))
    int c0 = wave * 64;
    v8f acc[4];
    {
        v8f z = {0.f, 0.f, 0.f, 0.f, 0.f, 0.f, 0.f, 0.f};
        acc[0] = z; acc[1] = z; acc[2] = z; acc[3] = z;
    }
    for (int kc = 0; kc < 4096; kc += 32) {
        bf16x16 af = load_frag_a(S + kc, 4096, lane);
        #pragma unroll
        for (int t = 0; t < 4; ++t) {
            bf16x16 bb = load_frag_b(vmat + ((size_t)b * 512 + c0 + t * 16) * 4096 + kc, 4096, lane);
            acc[t] = wmma_bf(af, bb, acc[t]);
        }
    }
    float alpha = alphap[0];
    #pragma unroll
    for (int t = 0; t < 4; ++t) {
        int c = c0 + t * 16 + nloc;
        size_t base = ((size_t)b * 512 + c) * 4096 + q0 + moff;
        union { uint4 q; unsigned short s[8]; } res, op;
        res.q = *(const uint4*)&xres[base];
        #pragma unroll
        for (int r = 0; r < 8; ++r) {
            float inv = 1.f / rowsum[moff + r];
            float val = alpha * (acc[t][r] * inv) + bits_bf16(res.s[r]);
            op.s[r] = bf16_bits(val);
        }
        *(uint4*)&out[base] = op.q;
    }
}

// ---------------------------------------------------------------------------
// CAM Gram matrix: G[b][i][j] = sum_p X[b][i][p] X[b][j][p]  (f32 out)
// Grid (8 jblk, 4 iblk, B); block 256 -> 128(i) x 64(j) tile.
// ---------------------------------------------------------------------------
__global__ __launch_bounds__(256)
void cam_gram_kernel(const bf16* __restrict__ X, float* __restrict__ G) {
    int b = blockIdx.z;
    int j0 = blockIdx.x * 64, i0 = blockIdx.y * 128;
    int tid = threadIdx.x, wave = tid >> 5, lane = tid & 31;
    int wi = (wave >> 1) * 32, wj = (wave & 1) * 32;
    v8f z = {0.f, 0.f, 0.f, 0.f, 0.f, 0.f, 0.f, 0.f};
    v8f acc[2][2];
    acc[0][0] = z; acc[0][1] = z; acc[1][0] = z; acc[1][1] = z;
    const bf16* Xb = X + (size_t)b * 512 * 4096;
    for (int kc = 0; kc < 4096; kc += 32) {
        bf16x16 a0 = load_frag_a(Xb + (size_t)(i0 + wi)      * 4096 + kc, 4096, lane);
        bf16x16 a1 = load_frag_a(Xb + (size_t)(i0 + wi + 16) * 4096 + kc, 4096, lane);
        bf16x16 b0 = load_frag_b(Xb + (size_t)(j0 + wj)      * 4096 + kc, 4096, lane);
        bf16x16 b1 = load_frag_b(Xb + (size_t)(j0 + wj + 16) * 4096 + kc, 4096, lane);
        acc[0][0] = wmma_bf(a0, b0, acc[0][0]);
        acc[0][1] = wmma_bf(a0, b1, acc[0][1]);
        acc[1][0] = wmma_bf(a1, b0, acc[1][0]);
        acc[1][1] = wmma_bf(a1, b1, acc[1][1]);
    }
    int nloc = lane & 15, moff = (lane < 16) ? 0 : 8;
    #pragma unroll
    for (int ti = 0; ti < 2; ++ti)
        #pragma unroll
        for (int tj = 0; tj < 2; ++tj)
            #pragma unroll
            for (int r = 0; r < 8; ++r) {
                int i = i0 + wi + ti * 16 + moff + r;
                int j = j0 + wj + tj * 16 + nloc;
                G[((size_t)b * 512 + i) * 512 + j] = acc[ti][tj][r];
            }
}

// softmax(minG - G) per row (== softmax(maxG - G) of the reference), bf16 out
__global__ void cam_softmax_kernel(const float* __restrict__ G, bf16* __restrict__ A) {
    __shared__ float red[256];
    int row = blockIdx.x;                       // b*512 + i
    const float* g = G + (size_t)row * 512;
    int t = threadIdx.x;
    float mn = fminf(g[t], g[t + 256]);
    red[t] = mn;
    __syncthreads();
    for (int s = 128; s > 0; s >>= 1) { if (t < s) red[t] = fminf(red[t], red[t + s]); __syncthreads(); }
    float minv = red[0];
    __syncthreads();
    float e0 = __expf(minv - g[t]);
    float e1 = __expf(minv - g[t + 256]);
    red[t] = e0 + e1;
    __syncthreads();
    for (int s = 128; s > 0; s >>= 1) { if (t < s) red[t] += red[t + s]; __syncthreads(); }
    float inv = 1.f / red[0];
    bf16* a = A + (size_t)row * 512;
    a[t] = (bf16)(e0 * inv);
    a[t + 256] = (bf16)(e1 * inv);
}

// ---------------------------------------------------------------------------
extern "C" void kernel_launch(void* const* d_in, const int* in_sizes, int n_in,
                              void* d_out, int out_size, void* d_ws, size_t ws_size,
                              hipStream_t stream) {
    (void)in_sizes; (void)n_in; (void)out_size; (void)ws_size;
    const int B = 2, Cin = 2048, Cm = 512, Cq = 64, HW = 4096, NC = 19;

    const float* x        = (const float*)d_in[0];
    const float* lfm      = (const float*)d_in[1];
    const float* wp1      = (const float*)d_in[2];
    const float* bnp1     = (const float*)d_in[3];
    const float* wp2      = (const float*)d_in[4];
    const float* bnp2     = (const float*)d_in[5];
    const float* wc1      = (const float*)d_in[6];
    const float* bnc1     = (const float*)d_in[7];
    const float* wc2      = (const float*)d_in[8];
    const float* bnc2     = (const float*)d_in[9];
    const float* wv1      = (const float*)d_in[10];
    const float* bnv1     = (const float*)d_in[11];
    const float* wv2      = (const float*)d_in[12];
    const float* bnv2     = (const float*)d_in[13];
    const float* pam_wb   = (const float*)d_in[14];
    const float* pam_bb   = (const float*)d_in[15];
    const float* pam_wc   = (const float*)d_in[16];
    const float* pam_bc   = (const float*)d_in[17];
    const float* pam_wd   = (const float*)d_in[18];
    const float* pam_bd   = (const float*)d_in[19];
    const float* pam_al   = (const float*)d_in[20];
    const float* cam_be   = (const float*)d_in[21];
    const float* vf_wb    = (const float*)d_in[22];
    const float* vf_bb    = (const float*)d_in[23];
    const float* vf_wc    = (const float*)d_in[24];
    const float* vf_bc    = (const float*)d_in[25];
    const float* vf_wd    = (const float*)d_in[26];
    const float* vf_bd    = (const float*)d_in[27];
    const float* vf_al    = (const float*)d_in[28];
    const float* out_w    = (const float*)d_in[29];
    const float* out_b    = (const float*)d_in[30];
    const float* p3_w     = (const float*)d_in[31];
    const float* p3_b     = (const float*)d_in[32];
    const float* c3_w     = (const float*)d_in[33];
    const float* c3_b     = (const float*)d_in[34];

    float* out = (float*)d_out;

    // --- workspace bump allocator ---
    char* ws = (char*)d_ws;
    size_t off = 0;
    auto alloc = [&](size_t bytes) -> void* {
        off = (off + 255) & ~(size_t)255;
        void* p = ws + off;
        off += bytes;
        return p;
    };
    const size_t actB = (size_t)B * Cm * HW * 2;   // bf16 activation [B][512][4096]
    bf16* xb    = (bf16*)alloc((size_t)B * Cin * HW * 2);
    bf16* wp1p  = (bf16*)alloc((size_t)9 * Cm * Cin * 2);
    bf16* wc1p  = (bf16*)alloc((size_t)9 * Cm * Cin * 2);
    bf16* wv1p  = (bf16*)alloc((size_t)9 * Cm * Cin * 2);
    bf16* wp2p  = (bf16*)alloc((size_t)9 * Cm * Cm * 2);
    bf16* wc2p  = (bf16*)alloc((size_t)9 * Cm * Cm * 2);
    bf16* wv2p  = (bf16*)alloc((size_t)9 * Cm * Cm * 2);
    bf16* pwbp  = (bf16*)alloc((size_t)Cq * Cm * 2);
    bf16* pwcp  = (bf16*)alloc((size_t)Cq * Cm * 2);
    bf16* pwdp  = (bf16*)alloc((size_t)Cm * Cm * 2);
    bf16* vwbp  = (bf16*)alloc((size_t)Cq * Cm * 2);
    bf16* vwcp  = (bf16*)alloc((size_t)Cq * Cm * 2);
    bf16* vwdp  = (bf16*)alloc((size_t)Cm * Cm * 2);
    bf16* owp   = (bf16*)alloc((size_t)NC * Cm * 2);
    bf16* p3wp  = (bf16*)alloc((size_t)NC * Cm * 2);
    bf16* c3wp  = (bf16*)alloc((size_t)NC * Cm * 2);
    float* scp1 = (float*)alloc(Cm * 4); float* shp1 = (float*)alloc(Cm * 4);
    float* scp2 = (float*)alloc(Cm * 4); float* shp2 = (float*)alloc(Cm * 4);
    float* scc1 = (float*)alloc(Cm * 4); float* shc1 = (float*)alloc(Cm * 4);
    float* scc2 = (float*)alloc(Cm * 4); float* shc2 = (float*)alloc(Cm * 4);
    float* scv1 = (float*)alloc(Cm * 4); float* shv1 = (float*)alloc(Cm * 4);
    float* scv2 = (float*)alloc(Cm * 4); float* shv2 = (float*)alloc(Cm * 4);
    bf16* fp1   = (bf16*)alloc(actB);
    bf16* fc1   = (bf16*)alloc(actB);
    bf16* fv1   = (bf16*)alloc(actB);
    bf16* qbuf  = (bf16*)alloc((size_t)B * Cq * HW * 2);
    bf16* kbuf  = (bf16*)alloc((size_t)B * Cq * HW * 2);
    bf16* qT    = (bf16*)alloc((size_t)B * Cq * HW * 2);
    bf16* kT    = (bf16*)alloc((size_t)B * Cq * HW * 2);
    bf16* vbuf  = (bf16*)alloc(actB);
    bf16* attout= (bf16*)alloc(actB);             // reused pam/cam/vfam sequentially
    bf16* fp2   = (bf16*)alloc(actB);
    bf16* fc2   = (bf16*)alloc(actB);
    bf16* fv2   = (bf16*)alloc(actB);
    float* G    = (float*)alloc((size_t)B * Cm * Cm * 4);
    bf16* attC  = (bf16*)alloc((size_t)B * Cm * Cm * 2);
    bf16* fusion= (bf16*)alloc(actB);

    dim3 blk(256);
    // --- casts / packing / BN prep ---
    {
        long n = (long)B * Cin * HW;
        cast_kernel<<<dim3((unsigned)(n / 256)), blk, 0, stream>>>(x, xb, n);
    }
    auto packw = [&](const float* w, bf16* dst, int Co, int Ci, int taps) {
        long n = (long)taps * Co * Ci;
        pack_w_kernel<<<dim3((unsigned)((n + 255) / 256)), blk, 0, stream>>>(w, dst, Co, Ci, taps, n);
    };
    packw(wp1, wp1p, Cm, Cin, 9); packw(wc1, wc1p, Cm, Cin, 9); packw(wv1, wv1p, Cm, Cin, 9);
    packw(wp2, wp2p, Cm, Cm, 9);  packw(wc2, wc2p, Cm, Cm, 9);  packw(wv2, wv2p, Cm, Cm, 9);
    packw(pam_wb, pwbp, Cq, Cm, 1); packw(pam_wc, pwcp, Cq, Cm, 1); packw(pam_wd, pwdp, Cm, Cm, 1);
    packw(vf_wb, vwbp, Cq, Cm, 1);  packw(vf_wc, vwcp, Cq, Cm, 1);  packw(vf_wd, vwdp, Cm, Cm, 1);
    packw(out_w, owp, NC, Cm, 1);   packw(p3_w, p3wp, NC, Cm, 1);   packw(c3_w, c3wp, NC, Cm, 1);
    bn_prep_kernel<<<dim3(2), blk, 0, stream>>>(bnp1, scp1, shp1, Cm);
    bn_prep_kernel<<<dim3(2), blk, 0, stream>>>(bnp2, scp2, shp2, Cm);
    bn_prep_kernel<<<dim3(2), blk, 0, stream>>>(bnc1, scc1, shc1, Cm);
    bn_prep_kernel<<<dim3(2), blk, 0, stream>>>(bnc2, scc2, shc2, Cm);
    bn_prep_kernel<<<dim3(2), blk, 0, stream>>>(bnv1, scv1, shv1, Cm);
    bn_prep_kernel<<<dim3(2), blk, 0, stream>>>(bnv2, scv2, shv2, Cm);

    dim3 gConv(B * 64, 4);   // Co=512
    dim3 gSmall(B * 64, 1);  // Co<=128
    const size_t shAttn = (size_t)16 * 4096 * 2;

    // ================= P branch =================
    conv_gemm_kernel<9, EP_BN_RELU><<<gConv, blk, 0, stream>>>(
        wp1p, 0, xb, fp1, nullptr, scp1, shp1, nullptr, nullptr, nullptr, nullptr, Cm, Cin);
    conv_gemm_kernel<1, EP_BIAS><<<gSmall, blk, 0, stream>>>(
        pwbp, 0, fp1, qbuf, nullptr, nullptr, nullptr, pam_bb, nullptr, nullptr, nullptr, Cq, Cm);
    conv_gemm_kernel<1, EP_BIAS><<<gSmall, blk, 0, stream>>>(
        pwcp, 0, fp1, kbuf, nullptr, nullptr, nullptr, pam_bc, nullptr, nullptr, nullptr, Cq, Cm);
    conv_gemm_kernel<1, EP_BIAS><<<gConv, blk, 0, stream>>>(
        pwdp, 0, fp1, vbuf, nullptr, nullptr, nullptr, pam_bd, nullptr, nullptr, nullptr, Cm, Cm);
    transpose_kernel<<<dim3(128, 2, B), blk, 0, stream>>>(qbuf, qT, Cq, HW);
    transpose_kernel<<<dim3(128, 2, B), blk, 0, stream>>>(kbuf, kT, Cq, HW);
    pam_attention_kernel<<<dim3(B * 256), blk, shAttn, stream>>>(qT, kT, vbuf, fp1, attout, pam_al);
    conv_gemm_kernel<9, EP_BN_RELU><<<gConv, blk, 0, stream>>>(
        wp2p, 0, attout, fp2, nullptr, scp2, shp2, nullptr, nullptr, nullptr, nullptr, Cm, Cm);

    // ================= C branch =================
    conv_gemm_kernel<9, EP_BN_RELU><<<gConv, blk, 0, stream>>>(
        wc1p, 0, xb, fc1, nullptr, scc1, shc1, nullptr, nullptr, nullptr, nullptr, Cm, Cin);
    cam_gram_kernel<<<dim3(8, 4, B), blk, 0, stream>>>(fc1, G);
    cam_softmax_kernel<<<dim3(B * Cm), blk, 0, stream>>>(G, attC);
    conv_gemm_kernel<1, EP_RES_ALPHA><<<gConv, blk, 0, stream>>>(
        attC, (size_t)Cm * Cm, fc1, attout, nullptr, nullptr, nullptr, nullptr, nullptr, fc1, cam_be, Cm, Cm);
    conv_gemm_kernel<9, EP_BN_RELU><<<gConv, blk, 0, stream>>>(
        wc2p, 0, attout, fc2, nullptr, scc2, shc2, nullptr, nullptr, nullptr, nullptr, Cm, Cm);

    // ================= V branch (VFAM: q,k,v scaled by lfm) =================
    conv_gemm_kernel<9, EP_BN_RELU><<<gConv, blk, 0, stream>>>(
        wv1p, 0, xb, fv1, nullptr, scv1, shv1, nullptr, nullptr, nullptr, nullptr, Cm, Cin);
    conv_gemm_kernel<1, EP_BIAS_LFM><<<gSmall, blk, 0, stream>>>(
        vwbp, 0, fv1, qbuf, nullptr, nullptr, nullptr, vf_bb, lfm, nullptr, nullptr, Cq, Cm);
    conv_gemm_kernel<1, EP_BIAS_LFM><<<gSmall, blk, 0, stream>>>(
        vwcp, 0, fv1, kbuf, nullptr, nullptr, nullptr, vf_bc, lfm, nullptr, nullptr, Cq, Cm);
    conv_gemm_kernel<1, EP_BIAS_LFM><<<gConv, blk, 0, stream>>>(
        vwdp, 0, fv1, vbuf, nullptr, nullptr, nullptr, vf_bd, lfm, nullptr, nullptr, Cm, Cm);
    transpose_kernel<<<dim3(128, 2, B), blk, 0, stream>>>(qbuf, qT, Cq, HW);
    transpose_kernel<<<dim3(128, 2, B), blk, 0, stream>>>(kbuf, kT, Cq, HW);
    pam_attention_kernel<<<dim3(B * 256), blk, shAttn, stream>>>(qT, kT, vbuf, fv1, attout, vf_al);
    conv_gemm_kernel<9, EP_BN_RELU><<<gConv, blk, 0, stream>>>(
        wv2p, 0, attout, fv2, nullptr, scv2, shv2, nullptr, nullptr, nullptr, nullptr, Cm, Cm);

    // ================= fusion + heads =================
    {
        long n = (long)B * Cm * HW;
        add3_kernel<<<dim3((unsigned)(n / 256)), blk, 0, stream>>>(fp2, fc2, fv2, fusion, n);
    }
    const size_t outStride = (size_t)B * NC * HW;   // 155648
    conv_gemm_kernel<1, EP_BIAS_F32><<<gSmall, blk, 0, stream>>>(
        owp, 0, fusion, nullptr, out, nullptr, nullptr, out_b, nullptr, nullptr, nullptr, NC, Cm);
    conv_gemm_kernel<1, EP_BIAS_F32><<<gSmall, blk, 0, stream>>>(
        p3wp, 0, fp2, nullptr, out + outStride, nullptr, nullptr, p3_b, nullptr, nullptr, nullptr, NC, Cm);
    conv_gemm_kernel<1, EP_BIAS_F32><<<gSmall, blk, 0, stream>>>(
        c3wp, 0, fc2, nullptr, out + 2 * outStride, nullptr, nullptr, c3_b, nullptr, nullptr, nullptr, NC, Cm);
}
